// DAFMoELayer_65472481460435
// MI455X (gfx1250) — compile-verified, matrix operand order checked by reading, not run on hardware
//
#include <hip/hip_runtime.h>
#include <hip/hip_bf16.h>
#include <math.h>

// Problem constants (match reference)
#define D_EMB  256
#define D_FF   512
#define N_EXP  8
#define N_CAT  1000
#define N_TOK  16384      // B*S = 8*2048
#define TILE_M 64         // tokens per block
#define CHUNK_F 64        // DF chunk kept in LDS
#define N_CHUNK (D_FF / CHUNK_F)          // 8 chunks per expert
#define N_CC    (N_EXP * N_CHUNK)         // 64 chunk-loads total

// Plain ext-vector types (union/ldst safe, unlike HIP's class vector types)
typedef __attribute__((ext_vector_type(16))) __bf16        v16bf;
typedef __attribute__((ext_vector_type(8)))  float         v8f;
typedef __attribute__((ext_vector_type(4)))  unsigned int  v4u;
typedef __attribute__((ext_vector_type(2)))  unsigned int  v2u;
typedef __attribute__((ext_vector_type(4)))  float         v4f;
typedef __attribute__((ext_vector_type(8)))  int           i32x8;
typedef __attribute__((ext_vector_type(4)))  int           i32x4;

#if defined(__has_builtin)
#if __has_builtin(__builtin_amdgcn_tensor_load_to_lds) && \
    __has_builtin(__builtin_amdgcn_s_wait_tensorcnt)
#define HAVE_TDM 1
#endif
#endif
#ifndef HAVE_TDM
#define HAVE_TDM 0
#endif

// LDS layout (dynamic shared, byte offsets); double-buffered weight chunks
#define OFF_SA    0u           // [64][256] bf16 h tile          32 KB
#define OFF_SW1   32768u       // W1 chunk buf0/buf1 [64f][256d] 64 KB
#define OFF_SW2   98304u       // W2 chunk buf0/buf1 [256d][64f] 64 KB
#define W_BUF_SZ  32768u       // bytes per weight-chunk buffer
#define OFF_SC1   163840u      // [64 m][64 f] gelu out           8 KB
#define OFF_SR    172032u      // float[64]
#define OFF_SM    172288u      // float[64]
#define OFF_SG    172544u      // float[64*8]
#define OFF_SIDX  174592u      // int[64]
#define SMEM_BYTES 174848u

__device__ __forceinline__ unsigned short f2bf(float x) {
    // round-to-nearest-even fp32 -> bf16
    unsigned u = __float_as_uint(x);
    unsigned r = u + 0x7FFFu + ((u >> 16) & 1u);
    return (unsigned short)(r >> 16);
}

union FragU { v16bf v; v4u q[2]; };

// ISA 7.12.2 16-bit A/B fragment: lane holds row lc=lane&15; its 16 bf16
// elements are two contiguous 8-element K-runs at k0+half*8 and
// k0+16+half*8  ->  two ds_load_b128 from a K-contiguous LDS tile.
__device__ __forceinline__ v16bf load_frag_b16(const unsigned short* base,
                                               int row, int rowStride,
                                               int k0, int half) {
    const unsigned short* p = base + (size_t)row * rowStride + k0 + half * 8;
    FragU f;
    f.q[0] = *(const v4u*)(p);
    f.q[1] = *(const v4u*)(p + 16);
    return f.v;
}

#if HAVE_TDM
// Issue one TDM tensor_load_to_lds: 2-byte elements, 2-D tile
// (tile0 elems/row, tile1 rows, row stride stride0 elems) -> contiguous LDS.
// Descriptor packing per CDNA5 ISA section 8.3 / 8.4 (normal mode).
// This toolchain exposes the 6-arg builtin:
//   (u32x4 g0, i32x8 g1, i32x4 g2, i32x4 g3, i32x8 pad, i32 cpol)
__device__ __forceinline__ void tdm_issue(unsigned lds_off,
                                          const unsigned short* gsrc,
                                          unsigned tile0, unsigned tile1,
                                          unsigned stride0) {
    unsigned long long ga = (unsigned long long)(uintptr_t)(const void*)gsrc;
    v4u g0;
    g0.x = 1u;                                    // count=1, user descriptor
    g0.y = lds_off;                               // lds_addr (bytes)
    g0.z = (unsigned)(ga & 0xFFFFFFFFu);          // global_addr[31:0]
    g0.w = (unsigned)((ga >> 32) & 0x01FFFFFFu)   // global_addr[56:32]
         | (2u << 30);                            // type=2 ("image")
    i32x8 g1;
    g1[0] = (int)(1u << 16);                      // data_size=1 (2B), mask=0
    g1[1] = (int)((tile0 & 0xFFFFu) << 16);       // tensor_dim0[15:0] @63:48
    g1[2] = (int)(((tile0 >> 16) & 0xFFFFu)       // tensor_dim0[31:16] @79:64
         |  ((tile1 & 0xFFFFu) << 16));           // tensor_dim1[15:0] @95:80
    g1[3] = (int)(((tile1 >> 16) & 0xFFFFu)       // tensor_dim1[31:16] @111:96
         |  ((tile0 & 0xFFFFu) << 16));           // tile_dim0 @127:112
    g1[4] = (int)(tile1 & 0xFFFFu);               // tile_dim1 @143:128 (tile_dim2=0)
    g1[5] = (int)stride0;                         // tensor_dim0_stride[31:0] @191:160
    g1[6] = 0;                                    // stride0[47:32], dim1_stride[15:0]
    g1[7] = 0;
    i32x4 z4 = {};                                // groups 2/3 unused (<=2D)
    i32x8 z8 = {};
    __builtin_amdgcn_tensor_load_to_lds((v4u)g0, g1, z4, z4, z8, 0);
}
#endif

// ---------------------------------------------------------------------------
// Prep: transpose + fp32->bf16 both expert weight tensors into workspace.
//   w1t[e][f][d] = bf16(w1[e][d][f])   (GEMM1 B operand, K=d contiguous)
//   w2t[e][d][f] = bf16(w2[e][f][d])   (GEMM2 B operand, K=f contiguous)
// ---------------------------------------------------------------------------
__global__ void daf_prep_weights(const float* __restrict__ w1,
                                 const float* __restrict__ w2,
                                 unsigned short* __restrict__ w1t,
                                 unsigned short* __restrict__ w2t) {
    size_t i = (size_t)blockIdx.x * blockDim.x + threadIdx.x;  // < E*D*DF
    {
        int d = (int)(i % D_EMB);
        size_t t = i / D_EMB;
        int f = (int)(t % D_FF);
        int e = (int)(t / D_FF);
        w1t[i] = f2bf(w1[((size_t)e * D_EMB + d) * D_FF + f]);
    }
    {
        int f = (int)(i % D_FF);
        size_t t = i / D_FF;
        int d = (int)(t % D_EMB);
        int e = (int)(t / D_EMB);
        w2t[i] = f2bf(w2[((size_t)e * D_FF + f) * D_EMB + d]);
    }
}

// ---------------------------------------------------------------------------
// Fused MoE: block = 64 tokens x full D output; loop experts; chunk DF with
// TDM-double-buffered weight tiles overlapped against WMMA compute.
// ---------------------------------------------------------------------------
__global__ __launch_bounds__(256)
void daf_moe_fused(const float* __restrict__ h,
                   const float* __restrict__ gw,
                   const float* __restrict__ mu,
                   const float* __restrict__ r_j,
                   const float* __restrict__ fmask,
                   const unsigned short* __restrict__ w1t,
                   const unsigned short* __restrict__ w2t,
                   const float* __restrict__ wnum,
                   const float* __restrict__ bnum,
                   const float* __restrict__ cat_emb,
                   const float* __restrict__ steep,
                   const float* __restrict__ thr,
                   float* __restrict__ out) {
    extern __shared__ unsigned char smem[];
    unsigned short* sA  = (unsigned short*)(smem + OFF_SA);
    unsigned short* sC1 = (unsigned short*)(smem + OFF_SC1);
    float* sR   = (float*)(smem + OFF_SR);
    float* sM   = (float*)(smem + OFF_SM);
    float* sG   = (float*)(smem + OFF_SG);
    int*   sIdx = (int*)(smem + OFF_SIDX);

    const int tid  = threadIdx.x;
    const int lane = tid & 31;
    const int wv   = tid >> 5;       // 8 wave32s
    const int half = lane >> 4;
    const int lc   = lane & 15;
    const int n0   = blockIdx.x * TILE_M;

    // ---- stage h tile fp32 -> bf16 into LDS
    {
        int row = tid >> 2, seg = tid & 3;
        const v4f* src = (const v4f*)(h + (size_t)(n0 + row) * D_EMB + seg * 64);
        v2u* dst = (v2u*)(sA + row * D_EMB + seg * 64);
        #pragma unroll
        for (int i = 0; i < 16; ++i) {
            v4f v = src[i];
            v2u o;
            o.x = (unsigned)f2bf(v.x) | ((unsigned)f2bf(v.y) << 16);
            o.y = (unsigned)f2bf(v.z) | ((unsigned)f2bf(v.w) << 16);
            dst[i] = o;
        }
    }
    // ---- per-token scalars
    if (tid < TILE_M) {
        float r = r_j[n0 + tid];
        float m = fmask[n0 + tid];
        sR[tid] = r;
        sM[tid] = m;
        sIdx[tid] = (int)(r * (1.0f - m));
    }
    for (int i = tid; i < TILE_M * N_EXP; i += 256)
        sG[i] = gw[(size_t)n0 * N_EXP + i];

    const int m_tile = wv & 3;            // output M tile of this wave
    const int d_base = (wv >> 2) * 8;     // 8 D tiles per wave
    const int g1_mt  = wv >> 1;           // GEMM1: shared M tile for 2 F tiles
    const int g1_ftb = (wv & 1) * 2;      // GEMM1: first of 2 F tiles

    v8f outAcc[8];
    #pragma unroll
    for (int i = 0; i < 8; ++i) outAcc[i] = (v8f){};

#if HAVE_TDM
    // prologue: DMA chunk 0 into buffer 0 (wave 0 owns TDM + TENSORcnt)
    if (wv == 0) {
        tdm_issue(OFF_SW1, w1t, CHUNK_F * D_EMB, 1, CHUNK_F * D_EMB);
        tdm_issue(OFF_SW2, w2t, CHUNK_F, D_EMB, D_FF);
    }
#endif

    for (int e = 0; e < N_EXP; ++e) {
        v8f eAcc[8];
        #pragma unroll
        for (int i = 0; i < 8; ++i) eAcc[i] = (v8f){};

        for (int fi = 0; fi < N_CHUNK; ++fi) {
            const int cc  = e * N_CHUNK + fi;
            const unsigned buf = (unsigned)(cc & 1);
            const unsigned short* sW1 =
                (const unsigned short*)(smem + OFF_SW1 + buf * W_BUF_SZ);
            const unsigned short* sW2 =
                (const unsigned short*)(smem + OFF_SW2 + buf * W_BUF_SZ);

#if HAVE_TDM
            if (wv == 0) {
                if (cc + 1 < N_CC) {
                    // prefetch next chunk into the other buffer (its last
                    // reader finished before the end-of-iteration barrier)
                    const int ne = (cc + 1) >> 3, nf = ((cc + 1) & 7) * CHUNK_F;
                    const unsigned nb = buf ^ 1u;
                    tdm_issue(OFF_SW1 + nb * W_BUF_SZ,
                              w1t + ((size_t)ne * D_FF + nf) * D_EMB,
                              CHUNK_F * D_EMB, 1, CHUNK_F * D_EMB);
                    tdm_issue(OFF_SW2 + nb * W_BUF_SZ,
                              w2t + (size_t)ne * D_EMB * D_FF + nf,
                              CHUNK_F, D_EMB, D_FF);
                    // TDM completes in order per wave: <=2 outstanding means
                    // chunk cc's pair has landed in LDS.
                    __builtin_amdgcn_s_wait_tensorcnt(2);
                } else {
                    __builtin_amdgcn_s_wait_tensorcnt(0);
                }
            }
#else
            // fallback: synchronous all-thread copy of chunk cc
            {
                const int f0 = fi * CHUNK_F;
                const v4u* src = (const v4u*)(w1t + ((size_t)e * D_FF + f0) * D_EMB);
                v4u* dst = (v4u*)(smem + OFF_SW1 + buf * W_BUF_SZ);
                #pragma unroll
                for (int i = 0; i < (CHUNK_F * D_EMB) / (8 * 256); ++i)
                    dst[tid + i * 256] = src[tid + i * 256];
                const v4u* s2 = (const v4u*)(w2t + ((size_t)e * D_EMB + tid) * D_FF + f0);
                v4u* d2 = (v4u*)((unsigned short*)(smem + OFF_SW2 + buf * W_BUF_SZ)
                                 + tid * CHUNK_F);
                #pragma unroll
                for (int i = 0; i < CHUNK_F / 8; ++i) d2[i] = s2[i];
            }
#endif
            __syncthreads();   // chunk cc resident for all waves

            // ---- GEMM1: C1[64,64] = sA x W1chunk ; one A-frag feeds 2 tiles
            {
                v8f acc0 = (v8f){}, acc1 = (v8f){};
                #pragma unroll
                for (int kk = 0; kk < D_EMB / 32; ++kk) {
                    v16bf a  = load_frag_b16(sA, g1_mt * 16 + lc, D_EMB, kk * 32, half);
                    v16bf b0 = load_frag_b16(sW1, (g1_ftb + 0) * 16 + lc, D_EMB, kk * 32, half);
                    v16bf b1 = load_frag_b16(sW1, (g1_ftb + 1) * 16 + lc, D_EMB, kk * 32, half);
                    acc0 = __builtin_amdgcn_wmma_f32_16x16x32_bf16(
                               false, a, false, b0, (short)0, acc0, false, false);
                    acc1 = __builtin_amdgcn_wmma_f32_16x16x32_bf16(
                               false, a, false, b1, (short)0, acc1, false, false);
                }
                // exact GELU -> bf16 -> sC1[m][f]
                #pragma unroll
                for (int j = 0; j < 8; ++j) {
                    int m = g1_mt * 16 + j + half * 8;
                    float x0 = acc0[j];
                    float x1 = acc1[j];
                    float g0 = 0.5f * x0 * (1.0f + erff(x0 * 0.70710678118654752f));
                    float g1 = 0.5f * x1 * (1.0f + erff(x1 * 0.70710678118654752f));
                    sC1[m * CHUNK_F + (g1_ftb + 0) * 16 + lc] = f2bf(g0);
                    sC1[m * CHUNK_F + (g1_ftb + 1) * 16 + lc] = f2bf(g1);
                }
            }
            __syncthreads();

            // ---- GEMM2 accumulate: eAcc += C1chunk x W2chunk
            #pragma unroll
            for (int kk = 0; kk < CHUNK_F / 32; ++kk) {
                v16bf a = load_frag_b16(sC1, m_tile * 16 + lc, CHUNK_F, kk * 32, half);
                #pragma unroll
                for (int i = 0; i < 8; ++i) {
                    v16bf b = load_frag_b16(sW2, (d_base + i) * 16 + lc, CHUNK_F, kk * 32, half);
                    eAcc[i] = __builtin_amdgcn_wmma_f32_16x16x32_bf16(
                                  false, a, false, b, (short)0, eAcc[i], false, false);
                }
            }
            __syncthreads();   // all waves done with buffer cc / sC1
        }

        // ---- epilogue for expert e: preservation paths + gating -> outAcc
        float mu_e  = mu[e];
        float sg    = 1.0f / (1.0f + expf(-mu_e));
        float dist  = fabsf(sg - 0.5f);
        float alpha = 1.0f / (1.0f + expf(-(steep[e] * (dist - thr[e]))));
        #pragma unroll
        for (int i = 0; i < 8; ++i) {
            int d = (d_base + i) * 16 + lc;
            float wn = wnum[e * D_EMB + d];
            float bn = bnum[e * D_EMB + d];
            const float* ce = cat_emb + (size_t)e * N_CAT * D_EMB + d;
            #pragma unroll
            for (int j = 0; j < 8; ++j) {
                int mrow = m_tile * 16 + j + half * 8;
                float rr = sR[mrow];
                float mm = sM[mrow];
                float cm = 1.0f - mm;
                float pres = mm * tanhf(rr * mm * wn + bn)
                           + cm * ce[(size_t)sIdx[mrow] * D_EMB];
                float g = sG[mrow * N_EXP + e];
                outAcc[i][j] += g * (eAcc[i][j] + alpha * pres);
            }
        }
    }

    // ---- write final output [N,D] f32
    #pragma unroll
    for (int i = 0; i < 8; ++i) {
        int d = (d_base + i) * 16 + lc;
        #pragma unroll
        for (int j = 0; j < 8; ++j) {
            int mrow = m_tile * 16 + j + half * 8;
            out[(size_t)(n0 + mrow) * D_EMB + d] = outAcc[i][j];
        }
    }
}

extern "C" void kernel_launch(void* const* d_in, const int* in_sizes, int n_in,
                              void* d_out, int out_size, void* d_ws, size_t ws_size,
                              hipStream_t stream) {
    const float* h    = (const float*)d_in[0];
    const float* gw   = (const float*)d_in[1];
    // d_in[2] selected_indices: unused by reference (dense all-expert FFN)
    const float* mu   = (const float*)d_in[3];
    const float* r_j  = (const float*)d_in[4];
    const float* fm   = (const float*)d_in[5];
    const float* w1   = (const float*)d_in[6];
    const float* w2   = (const float*)d_in[7];
    const float* wnum = (const float*)d_in[8];
    const float* bnum = (const float*)d_in[9];
    const float* cemb = (const float*)d_in[10];
    const float* stp  = (const float*)d_in[11];
    const float* thr  = (const float*)d_in[12];
    float* out = (float*)d_out;

    unsigned short* w1t = (unsigned short*)d_ws;                      // 2 MB
    unsigned short* w2t = w1t + (size_t)N_EXP * D_FF * D_EMB;         // 2 MB

    daf_prep_weights<<<(N_EXP * D_EMB * D_FF) / 256, 256, 0, stream>>>(
        w1, w2, w1t, w2t);

    daf_moe_fused<<<N_TOK / TILE_M, 256, SMEM_BYTES, stream>>>(
        h, gw, mu, r_j, fm, w1t, w2t, wnum, bnum, cemb, stp, thr, out);
}
